// Llama2_7BRandomTransformEmbedder_6940667150918
// MI455X (gfx1250) — compile-verified, matrix operand order checked by reading, not run on hardware
//
#include <hip/hip_runtime.h>

// ---------------- model constants ----------------
#define D_      1024
#define FF_     2816
#define NL_     2
#define B_      2
#define L_      128
#define TNEW_   8
#define HOUT_   1124
#define VOC_    32000
#define NH_     16
#define HD_     64
#define MAXS_   136   // 128 + 8 generated tokens

typedef __attribute__((ext_vector_type(16))) __bf16          v16bf;
typedef __attribute__((ext_vector_type(16))) unsigned short  v16u;
typedef __attribute__((ext_vector_type(8)))  float           v8f;

static __device__ __forceinline__ unsigned short f32_to_bf16(float f) {
  unsigned int u = __float_as_uint(f);
  u += 0x7FFFu + ((u >> 16) & 1u);      // round-to-nearest-even
  return (unsigned short)(u >> 16);
}

// ---------------------------------------------------------------------------
// WMMA 16-lane x 32-k tile mapping (ISA 7.12.2, 16-bit operands).
// For the "16-dim" index p (row of A / col of B) and k in [0,32):
//   k in [0,8):   lane = p,      elem = k
//   k in [8,16):  lane = p + 16, elem = k - 8
//   k in [16,24): lane = p,      elem = 8 + (k - 16)
//   k in [24,32): lane = p + 16, elem = 8 + (k - 24)
// Tiles stored (16dim-tile)-major then kt; lane-major inside (32 x 16 bf16).
// ---------------------------------------------------------------------------

// Pack fp32 weight W[K x N] (row-major) -> bf16 B-layout tiles. grid=(N/256, K).
__global__ void pack_w_bf16(const float* __restrict__ W, unsigned short* __restrict__ Wp,
                            int N) {
  int n = blockIdx.x * blockDim.x + threadIdx.x;
  int k = blockIdx.y;
  int K = gridDim.y;
  if (n >= N) return;
  int kt = k >> 5, kk = k & 31;
  int nt = n >> 4, nn = n & 15;
  int seg = kk >> 3;
  int off = kk & 7;
  int lane = nn + ((seg & 1) ? 16 : 0);
  int pos  = off + ((seg >> 1) ? 8 : 0);
  int ktiles = K >> 5;
  size_t dst = ((((size_t)nt * ktiles + kt) * 32) + lane) * 16 + pos;
  Wp[dst] = f32_to_bf16(W[(size_t)k * N + n]);
}

// Pack fp32 activations A[M x K] (row-major) -> bf16 A-layout tiles,
// zero-padding rows m >= M up to Mpad. grid=(K/256, Mpad).
__global__ void pack_a_bf16(const float* __restrict__ A, unsigned short* __restrict__ Ap,
                            int M, int K) {
  int k = blockIdx.x * blockDim.x + threadIdx.x;
  int m = blockIdx.y;
  if (k >= K) return;
  float v = (m < M) ? A[(size_t)m * K + k] : 0.f;
  int kt = k >> 5, kk = k & 31;
  int mt = m >> 4, mm = m & 15;
  int seg = kk >> 3;
  int off = kk & 7;
  int lane = mm + ((seg & 1) ? 16 : 0);
  int pos  = off + ((seg >> 1) ? 8 : 0);
  int ktiles = K >> 5;
  size_t dst = ((((size_t)mt * ktiles + kt) * 32) + lane) * 16 + pos;
  Ap[dst] = f32_to_bf16(v);
}

// ---------------------------------------------------------------------------
// C[16 x 64 tile] = Ap(tile row) x Wp(4 column streams), K stepped 64/iter.
// One wave per 16x64 C tile; 8 WMMAs per loop body; no branches in loop.
// grid = (Mpad/16, N/64), block = 32.
// ---------------------------------------------------------------------------
__global__ __launch_bounds__(32) void gemm_bf16_wmma(
    const unsigned short* __restrict__ Ap,
    const unsigned short* __restrict__ Wp,
    float* __restrict__ C, int ldc, int K) {
  const int lane = threadIdx.x;
  const int tM  = blockIdx.x;
  const int tN4 = blockIdx.y;
  const int ktiles = K >> 5;
  const size_t stream = (size_t)ktiles * 32;   // v16u elements per tile-stream

  const v16u* ap  = (const v16u*)Ap + (size_t)tM * stream + lane;
  const v16u* wp0 = (const v16u*)Wp + (size_t)(tN4 * 4) * stream + lane;
  const v16u* wp1 = wp0 + stream;
  const v16u* wp2 = wp1 + stream;
  const v16u* wp3 = wp2 + stream;

  v8f c0 = {0.f,0.f,0.f,0.f,0.f,0.f,0.f,0.f};
  v8f c1 = c0, c2 = c0, c3 = c0;

  for (int kt = 0; kt < ktiles; kt += 2) {
    __builtin_prefetch(ap + 128, 0, 1);
    __builtin_prefetch(wp0 + 128, 0, 1);
    v16bf a0  = __builtin_bit_cast(v16bf, ap[0]);
    v16bf a1  = __builtin_bit_cast(v16bf, ap[32]);
    v16bf b00 = __builtin_bit_cast(v16bf, wp0[0]);
    v16bf b01 = __builtin_bit_cast(v16bf, wp0[32]);
    v16bf b10 = __builtin_bit_cast(v16bf, wp1[0]);
    v16bf b11 = __builtin_bit_cast(v16bf, wp1[32]);
    v16bf b20 = __builtin_bit_cast(v16bf, wp2[0]);
    v16bf b21 = __builtin_bit_cast(v16bf, wp2[32]);
    v16bf b30 = __builtin_bit_cast(v16bf, wp3[0]);
    v16bf b31 = __builtin_bit_cast(v16bf, wp3[32]);
    c0 = __builtin_amdgcn_wmma_f32_16x16x32_bf16(false, a0, false, b00, (short)0, c0, false, false);
    c1 = __builtin_amdgcn_wmma_f32_16x16x32_bf16(false, a0, false, b10, (short)0, c1, false, false);
    c2 = __builtin_amdgcn_wmma_f32_16x16x32_bf16(false, a0, false, b20, (short)0, c2, false, false);
    c3 = __builtin_amdgcn_wmma_f32_16x16x32_bf16(false, a0, false, b30, (short)0, c3, false, false);
    c0 = __builtin_amdgcn_wmma_f32_16x16x32_bf16(false, a1, false, b01, (short)0, c0, false, false);
    c1 = __builtin_amdgcn_wmma_f32_16x16x32_bf16(false, a1, false, b11, (short)0, c1, false, false);
    c2 = __builtin_amdgcn_wmma_f32_16x16x32_bf16(false, a1, false, b21, (short)0, c2, false, false);
    c3 = __builtin_amdgcn_wmma_f32_16x16x32_bf16(false, a1, false, b31, (short)0, c3, false, false);
    ap += 64; wp0 += 64; wp1 += 64; wp2 += 64; wp3 += 64;
  }

  const int col   = tN4 * 64 + (lane & 15);
  const int rbase = tM * 16 + ((lane >> 4) << 3);
#pragma unroll
  for (int j = 0; j < 8; ++j) {
    float* crow = C + (size_t)(rbase + j) * ldc + col;
    crow[0]  = c0[j];
    crow[16] = c1[j];
    crow[32] = c2[j];
    crow[48] = c3[j];
  }
}

// ---------------------------------------------------------------------------
__global__ __launch_bounds__(256) void rmsnorm_kernel(
    const float* __restrict__ in, size_t in_stride,
    const float* __restrict__ w,
    float* __restrict__ outp, size_t out_stride) {
  int row = blockIdx.x;
  const float* x = in + (size_t)row * in_stride;
  float* y = outp + (size_t)row * out_stride;
  __shared__ float red[8];
  float s = 0.f;
  for (int i = threadIdx.x; i < D_; i += 256) { float v = x[i]; s += v * v; }
#pragma unroll
  for (int o = 16; o > 0; o >>= 1) s += __shfl_down(s, o, 32);
  if ((threadIdx.x & 31) == 0) red[threadIdx.x >> 5] = s;
  __syncthreads();
  if (threadIdx.x == 0) {
    float t = 0.f;
    for (int i = 0; i < 8; ++i) t += red[i];
    red[0] = rsqrtf(t / (float)D_ + 1e-5f);
  }
  __syncthreads();
  float r = red[0];
  for (int i = threadIdx.x; i < D_; i += 256) y[i] = x[i] * r * w[i];
}

// rope applied in place to q (compact rows, b-major)
__global__ void rope_q_kernel(float* __restrict__ q, int nq, int p0, int total) {
  int i = blockIdx.x * blockDim.x + threadIdx.x;
  if (i >= total) return;
  int hi = i & 31;
  int h  = (i >> 5) & 15;
  int row = i >> 9;
  int pos = p0 + (row % nq);
  float inv = __powf(10000.0f, -(float)hi / 32.0f);
  float ang = (float)pos * inv;
  float c = __cosf(ang), s = __sinf(ang);
  float* base = q + (size_t)row * D_ + h * HD_;
  float x1 = base[hi], x2 = base[hi + 32];
  base[hi]      = x1 * c - x2 * s;
  base[hi + 32] = x1 * s + x2 * c;
}

// rope(k) -> K cache; v -> V cache
__global__ void kv_cache_kernel(const float* __restrict__ kc, const float* __restrict__ vc,
                                float* __restrict__ Kc, float* __restrict__ Vc,
                                int nq, int p0, int total) {
  int i = blockIdx.x * blockDim.x + threadIdx.x;
  if (i >= total) return;
  int hi = i & 31;
  int h  = (i >> 5) & 15;
  int row = i >> 9;
  int b = row / nq, j = row % nq, pos = p0 + j;
  size_t src = (size_t)row * D_ + h * HD_;
  size_t dst = ((size_t)(b * MAXS_ + pos)) * D_ + h * HD_;
  float x1 = kc[src + hi], x2 = kc[src + hi + 32];
  float inv = __powf(10000.0f, -(float)hi / 32.0f);
  float ang = (float)pos * inv;
  float c = __cosf(ang), s = __sinf(ang);
  Kc[dst + hi]      = x1 * c - x2 * s;
  Kc[dst + hi + 32] = x1 * s + x2 * c;
  Vc[dst + hi]      = vc[src + hi];
  Vc[dst + hi + 32] = vc[src + hi + 32];
}

// one block per (q position, head, batch); 64 threads
__global__ __launch_bounds__(64) void attn_kernel(
    const float* __restrict__ q, const float* __restrict__ Kc,
    const float* __restrict__ Vc, float* __restrict__ o, int nq, int p0) {
  int j = blockIdx.x, h = blockIdx.y, b = blockIdx.z;
  int t = threadIdx.x;
  int qpos = p0 + j;
  int row = b * nq + j;
  __shared__ float qs[HD_];
  __shared__ float sc[MAXS_];
  __shared__ float rinv;
  qs[t] = q[(size_t)row * D_ + h * HD_ + t];
  __syncthreads();
  int nk = qpos + 1;
  const float* Kb = Kc + ((size_t)b * MAXS_) * D_ + h * HD_;
  for (int k = t; k < nk; k += 64) {
    const float* kv = Kb + (size_t)k * D_;
    float s = 0.f;
#pragma unroll
    for (int d = 0; d < HD_; ++d) s += qs[d] * kv[d];
    sc[k] = s * 0.125f;   // 1/sqrt(64)
  }
  __syncthreads();
  if (t == 0) {
    float m = -1e30f;
    for (int k = 0; k < nk; ++k) m = fmaxf(m, sc[k]);
    float sum = 0.f;
    for (int k = 0; k < nk; ++k) { float e = __expf(sc[k] - m); sc[k] = e; sum += e; }
    rinv = 1.0f / sum;
  }
  __syncthreads();
  float inv = rinv;
  const float* Vb = Vc + ((size_t)b * MAXS_) * D_ + h * HD_ + t;
  float acc = 0.f;
  for (int k = 0; k < nk; ++k) acc += sc[k] * Vb[(size_t)k * D_];
  o[(size_t)row * D_ + h * HD_ + t] = acc * inv;
}

__global__ void add_kernel(float* __restrict__ a, const float* __restrict__ b, int n) {
  int i = blockIdx.x * blockDim.x + threadIdx.x;
  if (i < n) a[i] += b[i];
}

__global__ void silu_mul_kernel(const float* __restrict__ g, const float* __restrict__ u,
                                float* outp, int n) {
  int i = blockIdx.x * blockDim.x + threadIdx.x;
  if (i < n) { float x = g[i]; outp[i] = (x / (1.f + __expf(-x))) * u[i]; }
}

__global__ void tokens_init_kernel(const int* __restrict__ ids, int* __restrict__ tokens, int n) {
  int i = blockIdx.x * blockDim.x + threadIdx.x;
  if (i >= n) return;
  int b = i / L_, s = i % L_;
  tokens[b * MAXS_ + s] = ids[i];
}

__global__ void embed_kernel(const int* __restrict__ tokens, const float* __restrict__ embed,
                             float* __restrict__ h, int nq, int p0, int total) {
  int i = blockIdx.x * blockDim.x + threadIdx.x;
  if (i >= total) return;
  int d = i & (D_ - 1);
  int row = i >> 10;
  int b = row / nq, pos = p0 + (row % nq);
  int tok = tokens[b * MAXS_ + pos];
  h[i] = embed[(size_t)tok * D_ + d];
}

// clr = lp - mean(lp) == logits - mean(logits)  (lse cancels); also argmax.
__global__ __launch_bounds__(256) void logits_post_kernel(
    const float* __restrict__ logits, float* __restrict__ outp,
    int* __restrict__ tokens, int step, int S) {
  int b = blockIdx.x;
  const float* lg = logits + (size_t)b * VOC_;
  __shared__ float ssum[8];
  __shared__ float smax[8];
  __shared__ int   sidx[8];
  float sum = 0.f, mx = -1e30f;
  int mi = 0;
  for (int v = threadIdx.x; v < VOC_; v += 256) {
    float x = lg[v];
    sum += x;
    if (x > mx) { mx = x; mi = v; }
  }
#pragma unroll
  for (int o = 16; o > 0; o >>= 1) {
    sum += __shfl_down(sum, o, 32);
    float omx = __shfl_down(mx, o, 32);
    int   omi = __shfl_down(mi, o, 32);
    if (omx > mx || (omx == mx && omi < mi)) { mx = omx; mi = omi; }
  }
  if ((threadIdx.x & 31) == 0) {
    int w = threadIdx.x >> 5;
    ssum[w] = sum; smax[w] = mx; sidx[w] = mi;
  }
  __syncthreads();
  if (threadIdx.x == 0) {
    float tsum = 0.f, tmx = -1e30f; int tmi = 0;
    for (int w = 0; w < 8; ++w) {
      tsum += ssum[w];
      if (smax[w] > tmx || (smax[w] == tmx && sidx[w] < tmi)) { tmx = smax[w]; tmi = sidx[w]; }
    }
    ssum[0] = tsum; sidx[0] = tmi;
  }
  __syncthreads();
  float mean = ssum[0] / (float)VOC_;
  for (int v = threadIdx.x; v < HOUT_; v += 256)
    outp[((size_t)b * TNEW_ + step) * HOUT_ + v] = lg[v] - mean;
  if (threadIdx.x == 0) tokens[b * MAXS_ + S] = sidx[0];
}

// ---------------------------------------------------------------------------
extern "C" void kernel_launch(void* const* d_in, const int* in_sizes, int n_in,
                              void* d_out, int out_size, void* d_ws, size_t ws_size,
                              hipStream_t stream) {
  const int*   input_ids = (const int*)d_in[0];
  const float* embed     = (const float*)d_in[1];
  const float* wq   = (const float*)d_in[2];
  const float* wk   = (const float*)d_in[3];
  const float* wv   = (const float*)d_in[4];
  const float* wo   = (const float*)d_in[5];
  const float* wg   = (const float*)d_in[6];
  const float* wu   = (const float*)d_in[7];
  const float* wd   = (const float*)d_in[8];
  const float* anrm = (const float*)d_in[9];
  const float* fnrm = (const float*)d_in[10];
  const float* finaln = (const float*)d_in[11];
  const float* lmh  = (const float*)d_in[12];
  float* outp = (float*)d_out;
  (void)in_sizes; (void)n_in; (void)out_size; (void)ws_size;

  char* ws = (char*)d_ws;
  size_t off = 0;
  auto alloc = [&](size_t bytes) -> void* {
    off = (off + 255) & ~(size_t)255;
    void* p = ws + off;
    off += bytes;
    return p;
  };
  auto cdiv = [](size_t a, size_t b) -> unsigned { return (unsigned)((a + b - 1) / b); };

  const size_t dd = (size_t)D_ * D_;
  const size_t df = (size_t)D_ * FF_;
  const size_t fd = (size_t)FF_ * D_;

  unsigned short* Wqp = (unsigned short*)alloc(NL_ * dd * 2);
  unsigned short* Wkp = (unsigned short*)alloc(NL_ * dd * 2);
  unsigned short* Wvp = (unsigned short*)alloc(NL_ * dd * 2);
  unsigned short* Wop = (unsigned short*)alloc(NL_ * dd * 2);
  unsigned short* Wgp = (unsigned short*)alloc(NL_ * df * 2);
  unsigned short* Wup = (unsigned short*)alloc(NL_ * df * 2);
  unsigned short* Wdp = (unsigned short*)alloc(NL_ * fd * 2);
  unsigned short* Whp = (unsigned short*)alloc((size_t)D_ * VOC_ * 2);

  unsigned short* Ap  = (unsigned short*)alloc((size_t)B_ * L_ * FF_ * 2);  // packed activations

  int*   tokens = (int*)alloc((size_t)B_ * MAXS_ * 4);
  float* h      = (float*)alloc((size_t)B_ * L_ * D_ * 4);
  float* xb     = (float*)alloc((size_t)B_ * L_ * D_ * 4);
  float* qb     = (float*)alloc((size_t)B_ * L_ * D_ * 4);
  float* kb     = (float*)alloc((size_t)B_ * L_ * D_ * 4);
  float* vb     = (float*)alloc((size_t)B_ * L_ * D_ * 4);
  float* ob     = (float*)alloc((size_t)B_ * L_ * D_ * 4);
  float* tb     = (float*)alloc((size_t)B_ * L_ * D_ * 4);
  float* gateb  = (float*)alloc((size_t)B_ * L_ * FF_ * 4);
  float* upb    = (float*)alloc((size_t)B_ * L_ * FF_ * 4);
  float* Kc     = (float*)alloc((size_t)NL_ * B_ * MAXS_ * D_ * 4);
  float* Vc     = (float*)alloc((size_t)NL_ * B_ * MAXS_ * D_ * 4);
  float* xlast  = (float*)alloc((size_t)B_ * D_ * 4);
  float* logits = (float*)alloc((size_t)16 * VOC_ * 4);   // 16 rows for padded stores

  // ---- pack all weights to bf16 WMMA-B layout (once per launch) ----
  for (int l = 0; l < NL_; ++l) {
    pack_w_bf16<<<dim3(D_ / 256, D_), 256, 0, stream>>>(wq + l * dd, Wqp + l * dd, D_);
    pack_w_bf16<<<dim3(D_ / 256, D_), 256, 0, stream>>>(wk + l * dd, Wkp + l * dd, D_);
    pack_w_bf16<<<dim3(D_ / 256, D_), 256, 0, stream>>>(wv + l * dd, Wvp + l * dd, D_);
    pack_w_bf16<<<dim3(D_ / 256, D_), 256, 0, stream>>>(wo + l * dd, Wop + l * dd, D_);
    pack_w_bf16<<<dim3(FF_ / 256, D_), 256, 0, stream>>>(wg + l * df, Wgp + l * df, FF_);
    pack_w_bf16<<<dim3(FF_ / 256, D_), 256, 0, stream>>>(wu + l * df, Wup + l * df, FF_);
    pack_w_bf16<<<dim3(D_ / 256, FF_), 256, 0, stream>>>(wd + l * fd, Wdp + l * fd, D_);
  }
  pack_w_bf16<<<dim3(VOC_ / 256, D_), 256, 0, stream>>>(lmh, Whp, VOC_);

  tokens_init_kernel<<<cdiv(B_ * L_, 256), 256, 0, stream>>>(input_ids, tokens, B_ * L_);

  // ---- prefill (step 0) then KV-cached decode (steps 1..7) ----
  for (int step = 0; step < TNEW_; ++step) {
    const int p0 = (step == 0) ? 0 : (L_ - 1 + step);
    const int nq = (step == 0) ? L_ : 1;
    const int M  = B_ * nq;
    const int Mpad = (M + 15) & ~15;
    const int Snew = p0 + nq;

    embed_kernel<<<cdiv((size_t)M * D_, 256), 256, 0, stream>>>(tokens, embed, h, nq, p0, M * D_);

    for (int l = 0; l < NL_; ++l) {
      float* KcL = Kc + (size_t)l * B_ * MAXS_ * D_;
      float* VcL = Vc + (size_t)l * B_ * MAXS_ * D_;

      // ---- attention ----
      rmsnorm_kernel<<<M, 256, 0, stream>>>(h, D_, anrm + (size_t)l * D_, xb, D_);
      pack_a_bf16<<<dim3(D_ / 256, Mpad), 256, 0, stream>>>(xb, Ap, M, D_);

      dim3 gq(Mpad / 16, D_ / 64);
      gemm_bf16_wmma<<<gq, 32, 0, stream>>>(Ap, Wqp + l * dd, qb, D_, D_);
      gemm_bf16_wmma<<<gq, 32, 0, stream>>>(Ap, Wkp + l * dd, kb, D_, D_);
      gemm_bf16_wmma<<<gq, 32, 0, stream>>>(Ap, Wvp + l * dd, vb, D_, D_);

      rope_q_kernel<<<cdiv((size_t)M * 512, 256), 256, 0, stream>>>(qb, nq, p0, M * 512);
      kv_cache_kernel<<<cdiv((size_t)M * 512, 256), 256, 0, stream>>>(kb, vb, KcL, VcL, nq, p0, M * 512);

      dim3 ga(nq, NH_, B_);
      attn_kernel<<<ga, 64, 0, stream>>>(qb, KcL, VcL, ob, nq, p0);

      pack_a_bf16<<<dim3(D_ / 256, Mpad), 256, 0, stream>>>(ob, Ap, M, D_);
      gemm_bf16_wmma<<<gq, 32, 0, stream>>>(Ap, Wop + l * dd, tb, D_, D_);
      add_kernel<<<cdiv((size_t)M * D_, 256), 256, 0, stream>>>(h, tb, M * D_);

      // ---- FFN ----
      rmsnorm_kernel<<<M, 256, 0, stream>>>(h, D_, fnrm + (size_t)l * D_, xb, D_);
      pack_a_bf16<<<dim3(D_ / 256, Mpad), 256, 0, stream>>>(xb, Ap, M, D_);

      dim3 gf(Mpad / 16, FF_ / 64);
      gemm_bf16_wmma<<<gf, 32, 0, stream>>>(Ap, Wgp + l * df, gateb, FF_, D_);
      gemm_bf16_wmma<<<gf, 32, 0, stream>>>(Ap, Wup + l * df, upb, FF_, D_);
      silu_mul_kernel<<<cdiv((size_t)M * FF_, 256), 256, 0, stream>>>(gateb, upb, gateb, M * FF_);

      pack_a_bf16<<<dim3(FF_ / 256, Mpad), 256, 0, stream>>>(gateb, Ap, M, FF_);
      gemm_bf16_wmma<<<gq, 32, 0, stream>>>(Ap, Wdp + l * fd, tb, D_, FF_);
      add_kernel<<<cdiv((size_t)M * D_, 256), 256, 0, stream>>>(h, tb, M * D_);
    }

    // ---- final norm on last position, lm_head, post ----
    rmsnorm_kernel<<<B_, 256, 0, stream>>>(h + (size_t)(nq - 1) * D_, (size_t)nq * D_, finaln, xlast, D_);
    pack_a_bf16<<<dim3(D_ / 256, 16), 256, 0, stream>>>(xlast, Ap, B_, D_);
    dim3 gl(1, VOC_ / 64);
    gemm_bf16_wmma<<<gl, 32, 0, stream>>>(Ap, Whp, logits, VOC_, D_);
    logits_post_kernel<<<B_, 256, 0, stream>>>(logits, outp, tokens, step, Snew);
  }
}
